// RecurrentAutoencoder_49684181680835
// MI455X (gfx1250) — compile-verified, matrix operand order checked by reading, not run on hardware
//
#include <hip/hip_runtime.h>
#include <hip/hip_bf16.h>

// ---------------------------------------------------------------------------
// Recurrent autoencoder: 6 stacked LSTM scans + output projection.
// B=512, T=336, F=1, E=128, H=256.
//
// MI455X strategy:
//  * Latency-bound recurrence -> exploit batch parallelism only: 32 work-
//    groups x 16 batch rows; each wave owns a 16-wide hidden slice and all
//    four gates, so cell state stays in f32 registers.
//  * gates = [x_t | h] @ [Wih|Whh]^T fused via v_wmma_f32_16x16x32_f16
//    (f16 operands, f32 accumulate); weights pre-swizzled once into the
//    native WMMA B lane layout -> one contiguous 32B load per gate per
//    K-step, L2-resident (~4MB << 192MB L2).
//  * Layer shapes are compile-time template params: K-loop fully unrolled,
//    A/B fragments explicitly double-buffered one K-step ahead to keep the
//    matrix pipes fed instead of stalling on s_wait_loadcnt 0.
// ---------------------------------------------------------------------------

typedef __attribute__((ext_vector_type(16))) _Float16 v16h;
typedef __attribute__((ext_vector_type(8)))  _Float16 v8h;
typedef __attribute__((ext_vector_type(8)))  float    v8f;

#define BATCH   512
#define SEQ_T   336
#define BT      16      // batch tile per workgroup
#define NTHREADS 512    // 16 waves (wave32)

#define MODE_IN_F32 1   // input tensor is f32 [B,T,1]
#define MODE_BCAST  2   // input is latent z [B,In], broadcast over time
#define MODE_LAST   4   // emit only final hidden state [B,H]

__device__ __forceinline__ float sigm_f(float x) {
    return 1.0f / (1.0f + __expf(-x));
}
__device__ __forceinline__ float tanh_f(float x) {
    float e = __expf(2.0f * x);
    return (e - 1.0f) / (e + 1.0f);
}

// ---------------------------------------------------------------------------
// Pack [Wih | Whh] (f32, gate-major rows [4H, In]+[4H, H]) into the per-lane
// WMMA B-operand layout (16-bit B 32x16 tiles), zero-padding K to Kpad.
// Packed index p = ((ntile*KT + kt)*32 + lane)*16 + j
//   lane<16: n=lane,    kbase=0 ; lane>=16: n=lane-16, kbase=8
//   j<8 : k = kt*32 + kbase + j ; j>=8: k = kt*32 + 16 + kbase + (j-8)
// ---------------------------------------------------------------------------
__global__ void pack_weights_kernel(const float* __restrict__ Wih,
                                    const float* __restrict__ Whh,
                                    _Float16* __restrict__ dst,
                                    int In, int In_pad, int Kpad, int H)
{
    const int KT = Kpad >> 5;
    const size_t total = (size_t)(4 * H) * (size_t)Kpad;
    size_t p = (size_t)blockIdx.x * blockDim.x + threadIdx.x;
    if (p >= total) return;

    int j    = (int)(p & 15);
    int lane = (int)((p >> 4) & 31);
    size_t tile = p >> 9;
    int kt    = (int)(tile % (size_t)KT);
    int ntile = (int)(tile / (size_t)KT);

    int n  = ntile * 16 + (lane & 15);
    int kb = (lane < 16) ? 0 : 8;
    int k  = kt * 32 + ((j < 8) ? (kb + j) : (16 + kb + (j - 8)));

    float v = 0.0f;
    if (k < In_pad) {
        if (k < In) v = Wih[(size_t)n * In + k];
    } else {
        v = Whh[(size_t)n * H + (k - In_pad)];
    }
    dst[p] = (_Float16)v;
}

// ---------------------------------------------------------------------------
// One LSTM layer, full time scan. Grid = B/BT workgroups, 512 threads.
// Wave w (w < H/16) computes hidden columns [16w, 16w+16) for all 16 batch
// rows: 4 gate accumulator chains (i,f,g,o) via WMMA over K = InPad + H.
// ---------------------------------------------------------------------------
template<int In, int InPad, int Kpad, int H, int MODE>
__global__ __launch_bounds__(NTHREADS)
void lstm_scan_kernel(const void* __restrict__ Xin,
                      const _Float16* __restrict__ Bpk,
                      const float* __restrict__ bias,
                      _Float16* __restrict__ Yout, int T)
{
    constexpr int KT  = Kpad / 32;          // K-steps per gate
    constexpr int NT4 = H / 16;             // 16-col tiles per gate
    constexpr size_t GSTRIDE = (size_t)NT4 * KT * 512; // halves between gates

    __shared__ _Float16 smem[BT * Kpad];    // A operand: [BT, Kpad] halves

    const int tid  = threadIdx.x;
    const int lane = tid & 31;
    const int w    = tid >> 5;
    const int b0   = blockIdx.x * BT;
    const bool active = (w < NT4);

    // zero A buffer (covers K zero-padding and h(t=-1)=0)
    for (int i = tid; i < BT * Kpad; i += NTHREADS) smem[i] = (_Float16)0.0f;

    // hoist per-lane gate biases
    float bi_ = 0.f, bf_ = 0.f, bg_ = 0.f, bo_ = 0.f;
    if (active) {
        int nc = w * 16 + (lane & 15);
        bi_ = bias[0 * H + nc];
        bf_ = bias[1 * H + nc];
        bg_ = bias[2 * H + nc];
        bo_ = bias[3 * H + nc];
    }

    // broadcast latent: load x region once
    if constexpr (MODE & MODE_BCAST) {
        const _Float16* Z = (const _Float16*)Xin;
        constexpr int ch = In / 8;
        for (int i = tid; i < BT * ch; i += NTHREADS) {
            int row = i / ch, c8 = i - row * ch;
            *(v8h*)&smem[row * Kpad + c8 * 8] =
                *(const v8h*)&Z[(size_t)(b0 + row) * In + c8 * 8];
        }
    }

    v8f cst = {};                            // cell state (f32, registers)
    const int arow  = lane & 15;             // A-fragment source row
    const int kbase = (lane < 16) ? 0 : 8;   // A/B lane K-phase
    const int mrow  = (lane < 16) ? 0 : 8;   // D-fragment row base

    const _Float16* arp   = &smem[arow * Kpad];
    const _Float16* bbase = Bpk + ((size_t)w * KT) * 512 + (size_t)lane * 16;

    auto loadA = [&](int kt) -> v16h {
        v8h lo = *(const v8h*)(arp + kt * 32 + kbase);
        v8h hi = *(const v8h*)(arp + kt * 32 + kbase + 16);
        return __builtin_shufflevector(lo, hi,
                0,1,2,3,4,5,6,7,8,9,10,11,12,13,14,15);
    };
    auto loadB = [&](int kt, int g) -> v16h {
        return *(const v16h*)(bbase + (size_t)g * GSTRIDE + (size_t)kt * 512);
    };

    for (int t = 0; t < T; ++t) {
        // ---- stage x_t into LDS -------------------------------------------
        if constexpr (MODE & MODE_IN_F32) {  // F=1 f32 input
            if (tid < BT) {
                const float* X = (const float*)Xin;
                smem[tid * Kpad] = (_Float16)X[(size_t)(b0 + tid) * T + t];
            }
        } else if constexpr (!(MODE & MODE_BCAST)) { // f16 sequence input
            const _Float16* X = (const _Float16*)Xin;
            constexpr int ch = In / 8;
            for (int i = tid; i < BT * ch; i += NTHREADS) {
                int row = i / ch, c8 = i - row * ch;
                *(v8h*)&smem[row * Kpad + c8 * 8] =
                    *(const v8h*)&X[((size_t)(b0 + row) * T + t) * (size_t)In + c8 * 8];
            }
        }
        __syncthreads();                     // x_t and h_{t-1} visible

        float hreg[8];
        if (active) {
            v8f ai = {}, af = {}, ag = {}, ao = {};
            // software pipeline: fragments for K-step kt+1 load while the
            // matrix pipes chew on kt (4 independent accumulator chains)
            v16h a_c  = loadA(0);
            v16h b0c = loadB(0, 0), b1c = loadB(0, 1);
            v16h b2c = loadB(0, 2), b3c = loadB(0, 3);
            #pragma unroll
            for (int kt = 0; kt < KT; ++kt) {
                v16h a_n, b0n, b1n, b2n, b3n;
                if (kt + 1 < KT) {
                    a_n  = loadA(kt + 1);
                    b0n = loadB(kt + 1, 0); b1n = loadB(kt + 1, 1);
                    b2n = loadB(kt + 1, 2); b3n = loadB(kt + 1, 3);
                    if (kt + 2 < KT)         // keep the weight stream warm
                        __builtin_prefetch(bbase + (size_t)(kt + 2) * 512, 0, 3);
                }
                ai = __builtin_amdgcn_wmma_f32_16x16x32_f16(false, a_c, false, b0c, (short)0, ai, false, false);
                af = __builtin_amdgcn_wmma_f32_16x16x32_f16(false, a_c, false, b1c, (short)0, af, false, false);
                ag = __builtin_amdgcn_wmma_f32_16x16x32_f16(false, a_c, false, b2c, (short)0, ag, false, false);
                ao = __builtin_amdgcn_wmma_f32_16x16x32_f16(false, a_c, false, b3c, (short)0, ao, false, false);
                if (kt + 1 < KT) {
                    a_c = a_n; b0c = b0n; b1c = b1n; b2c = b2n; b3c = b3n;
                }
            }
            #pragma unroll
            for (int r = 0; r < 8; ++r) {
                float iv = sigm_f(ai[r] + bi_);
                float fv = sigm_f(af[r] + bf_);
                float gv = tanh_f(ag[r] + bg_);
                float ov = sigm_f(ao[r] + bo_);
                float cv = fv * cst[r] + iv * gv;
                cst[r] = cv;
                hreg[r] = ov * tanh_f(cv);
            }
        }
        __syncthreads();                     // all A-operand reads done

        if (active) {
            #pragma unroll
            for (int r = 0; r < 8; ++r) {
                int m = mrow + r;            // D layout: lane<16 -> M=r else 8+r
                _Float16 hv = (_Float16)hreg[r];
                int nc = w * 16 + (lane & 15);
                smem[m * Kpad + InPad + nc] = hv;         // h for next step
                if constexpr (!(MODE & MODE_LAST)) {
                    Yout[((size_t)(b0 + m) * T + t) * (size_t)H + nc] = hv;
                } else {
                    if (t == T - 1)
                        Yout[(size_t)(b0 + m) * H + nc] = hv;  // latent z
                }
            }
        }
    }
}

// ---------------------------------------------------------------------------
// out[b,t] = dot(h3[b,t,:], Wout) + bout   (F == 1)
// One wave per row: each lane loads 8 contiguous f16 (one 16B dword4) and
// 8 f32 weights, then a wave32 shuffle reduction.
// ---------------------------------------------------------------------------
template<int HN>
__global__ __launch_bounds__(256)
void out_proj_kernel(const _Float16* __restrict__ Hin,
                     const float* __restrict__ Wout,
                     const float* __restrict__ bout,
                     float* __restrict__ out, long total)
{
    long row = (long)blockIdx.x * 8 + (threadIdx.x >> 5);
    int lane = threadIdx.x & 31;
    if (row >= total) return;
    static_assert(HN == 32 * 8, "one 8-wide chunk per lane");
    v8h hv = *(const v8h*)(Hin + row * (size_t)HN + lane * 8);
    float s = 0.0f;
    #pragma unroll
    for (int i = 0; i < 8; ++i) s += (float)hv[i] * Wout[lane * 8 + i];
    #pragma unroll
    for (int off = 16; off > 0; off >>= 1) s += __shfl_down(s, off, 32);
    if (lane == 0) out[row] = s + bout[0];
}

// ---------------------------------------------------------------------------
extern "C" void kernel_launch(void* const* d_in, const int* in_sizes, int n_in,
                              void* d_out, int out_size, void* d_ws, size_t ws_size,
                              hipStream_t stream)
{
    (void)in_sizes; (void)n_in; (void)out_size; (void)ws_size;

    const float* x      = (const float*)d_in[0];
    const float* Wih[6] = { (const float*)d_in[1],  (const float*)d_in[4],
                            (const float*)d_in[7],  (const float*)d_in[10],
                            (const float*)d_in[13], (const float*)d_in[16] };
    const float* Whh[6] = { (const float*)d_in[2],  (const float*)d_in[5],
                            (const float*)d_in[8],  (const float*)d_in[11],
                            (const float*)d_in[14], (const float*)d_in[17] };
    const float* bb[6]  = { (const float*)d_in[3],  (const float*)d_in[6],
                            (const float*)d_in[9],  (const float*)d_in[12],
                            (const float*)d_in[15], (const float*)d_in[18] };
    const float* Wout = (const float*)d_in[19];
    const float* bout = (const float*)d_in[20];

    // layer configs: In, In_pad, Kpad, H
    const int Ins[6]    = {   1, 256, 256, 128, 128, 256 };
    const int InPads[6] = {  32, 256, 256, 128, 128, 256 };
    const int Kpads[6]  = { 288, 512, 384, 256, 384, 512 };
    const int Hs[6]     = { 256, 256, 128, 128, 256, 256 };

    // workspace carve-out (halves, 256-half aligned)
    _Float16* ws = (_Float16*)d_ws;
    size_t off = 0;
    auto carve = [&](size_t n) -> _Float16* {
        _Float16* p = ws + off;
        off += (n + 255) & ~(size_t)255;
        return p;
    };
    _Float16* bpk[6];
    for (int l = 0; l < 6; ++l) bpk[l] = carve((size_t)4 * Hs[l] * Kpads[l]);
    _Float16* zlat = carve((size_t)BATCH * 128);
    _Float16* buf0 = carve((size_t)BATCH * SEQ_T * 256);
    _Float16* buf1 = carve((size_t)BATCH * SEQ_T * 256);

    // 1) pack weights into WMMA B-operand layout (f32 -> f16, K zero-padded)
    for (int l = 0; l < 6; ++l) {
        size_t total = (size_t)4 * Hs[l] * Kpads[l];
        int blocks = (int)((total + 255) / 256);
        pack_weights_kernel<<<blocks, 256, 0, stream>>>(
            Wih[l], Whh[l], bpk[l], Ins[l], InPads[l], Kpads[l], Hs[l]);
    }

    const int nblk = BATCH / BT;   // 32 workgroups, each owns 16 batch rows

    // 2) encoder
    lstm_scan_kernel<1, 32, 288, 256, MODE_IN_F32>
        <<<nblk, NTHREADS, 0, stream>>>(x,    bpk[0], bb[0], buf1, SEQ_T);
    lstm_scan_kernel<256, 256, 512, 256, 0>
        <<<nblk, NTHREADS, 0, stream>>>(buf1, bpk[1], bb[1], buf0, SEQ_T);
    lstm_scan_kernel<256, 256, 384, 128, MODE_LAST>
        <<<nblk, NTHREADS, 0, stream>>>(buf0, bpk[2], bb[2], zlat, SEQ_T);

    // 3) decoder (latent broadcast across time)
    lstm_scan_kernel<128, 128, 256, 128, MODE_BCAST>
        <<<nblk, NTHREADS, 0, stream>>>(zlat, bpk[3], bb[3], buf0, SEQ_T);
    lstm_scan_kernel<128, 128, 384, 256, 0>
        <<<nblk, NTHREADS, 0, stream>>>(buf0, bpk[4], bb[4], buf1, SEQ_T);
    lstm_scan_kernel<256, 256, 512, 256, 0>
        <<<nblk, NTHREADS, 0, stream>>>(buf1, bpk[5], bb[5], buf0, SEQ_T);

    // 4) output projection
    long total = (long)BATCH * SEQ_T;
    out_proj_kernel<256><<<(int)((total + 7) / 8), 256, 0, stream>>>(
        buf0, Wout, bout, (float*)d_out, total);
}
